// MultiHeadSelfAttention_88562225644082
// MI455X (gfx1250) — matrix operand order repaired and checked by reading, not compiled, not run on hardware
//
#include <hip/hip_runtime.h>
#include <hip/hip_bf16.h>
#include <stdint.h>

// ---------------------------------------------------------------------------
// MHSA forward for MI455X (gfx1250): bf16 WMMA everywhere, fp32 accumulate.
// B=2, S=2048, D_MODEL=1024, H=16, DK=64.
// GEMMs use 32x64 wave tiles (8 WMMA / K-step, 1.5 b128-loads per WMMA).
// ---------------------------------------------------------------------------

#define B_    2
#define S_    2048
#define D_    1024
#define H_    16
#define DK_   64
#define MROWS (B_ * S_)          // 4096

typedef __attribute__((ext_vector_type(16))) __bf16        v16bf;
typedef __attribute__((ext_vector_type(8)))  __bf16        v8bf;
typedef __attribute__((ext_vector_type(8)))  float         v8f;
typedef __attribute__((ext_vector_type(8)))  unsigned short v8u16;

static __device__ __forceinline__ unsigned short f2bf(float f) {
    unsigned u = __builtin_bit_cast(unsigned, f);
    u += 0x7FFFu + ((u >> 16) & 1u);           // round-to-nearest-even
    return (unsigned short)(u >> 16);
}

static __device__ __forceinline__ v16bf concat8(v8bf lo, v8bf hi) {
    return __builtin_shufflevector(lo, hi, 0,1,2,3,4,5,6,7,8,9,10,11,12,13,14,15);
}

static __device__ __forceinline__ v8f wmma_bf16(v16bf a, v16bf b, v8f c) {
    // D = A(16x32 bf16) x B(32x16 bf16) + C(16x16 f32)
    return __builtin_amdgcn_wmma_f32_16x16x32_bf16(
        /*neg_a=*/false, a, /*neg_b=*/false, b,
        /*c_mod=*/(short)0, c, /*reuse_a=*/false, /*reuse_b=*/false);
}

static __device__ __forceinline__ float rmax16(float v) {
    v = fmaxf(v, __shfl_xor(v, 1, 32));
    v = fmaxf(v, __shfl_xor(v, 2, 32));
    v = fmaxf(v, __shfl_xor(v, 4, 32));
    v = fmaxf(v, __shfl_xor(v, 8, 32));
    return v;
}
static __device__ __forceinline__ float rsum16(float v) {
    v += __shfl_xor(v, 1, 32);
    v += __shfl_xor(v, 2, 32);
    v += __shfl_xor(v, 4, 32);
    v += __shfl_xor(v, 8, 32);
    return v;
}

// ---------------------------------------------------------------------------
// K0: fp32 -> bf16 conversion
// ---------------------------------------------------------------------------
__global__ void mhsa_cvt_bf16(const float* __restrict__ src,
                              unsigned short* __restrict__ dst, int n) {
    for (int i = blockIdx.x * blockDim.x + threadIdx.x; i < n;
         i += gridDim.x * blockDim.x)
        dst[i] = f2bf(src[i]);
}

// ---------------------------------------------------------------------------
// K1: projection GEMM  Y[row,o] = sum_d X[row,d] * W[o,d]
//   mode 0/1 : Q/K -> RoPE epilogue, layout [b*H+h][s][dk]   (bf16)
//   mode 2   : V   -> transposed layout [b*H+h][dk][s]       (bf16)
// Wave tile: 32 rows x 64 cols (2 A-frags x 4 B-tiles), K-step 32.
// ---------------------------------------------------------------------------
__global__ void mhsa_proj(const unsigned short* __restrict__ Xb,   // [4096][1024]
                          const unsigned short* __restrict__ Wb,   // [1024][1024]
                          const int*            __restrict__ tp,   // [S]
                          unsigned short*       __restrict__ Out,
                          int mode) {
    const int lane    = threadIdx.x & 31;
    const int wid     = threadIdx.x >> 5;
    const int gw      = blockIdx.x * 8 + wid;       // 0..2047
    const int mBase   = (gw >> 4) * 32;             // 0..4064
    const int nBase   = (gw & 15) * 64;             // 0..960
    const int ln      = lane & 15;
    const int halfSel = lane >> 4;
    const int koffA   = halfSel * 16;
    const int koffB   = halfSel * 16;

    v8f acc[2][4] = {};
    const unsigned short* xrow0 = Xb + (size_t)(mBase + ln) * D_;
    const unsigned short* xrow1 = Xb + (size_t)(mBase + 16 + ln) * D_;

    for (int k = 0; k < D_; k += 32) {
        if (k + 64 < D_) {
            __builtin_prefetch(xrow0 + k + 64, 0, 0);
            __builtin_prefetch(xrow1 + k + 64, 0, 0);
        }
        v16bf a0 = concat8(*(const v8bf*)(xrow0 + k + koffA),
                           *(const v8bf*)(xrow0 + k + koffA + 8));
        v16bf a1 = concat8(*(const v8bf*)(xrow1 + k + koffA),
                           *(const v8bf*)(xrow1 + k + koffA + 8));
#pragma unroll
        for (int t = 0; t < 4; ++t) {
            const unsigned short* wrow = Wb + (size_t)(nBase + t * 16 + ln) * D_;
            v16bf b = *(const v16bf*)(wrow + k + koffB);
            acc[0][t] = wmma_bf16(a0, b, acc[0][t]);
            acc[1][t] = wmma_bf16(a1, b, acc[1][t]);
        }
    }

#pragma unroll
    for (int mi = 0; mi < 2; ++mi) {
        const int mB    = mBase + mi * 16;
        const int bb    = mB >> 11;                 // batch (tile stays in batch)
        const int sTile = mB & (S_ - 1);

        if (mode == 2) {
            // V: Vt[(b*H+h)*64+dk][2048]; 8 consecutive s per lane -> packed store
#pragma unroll
            for (int t = 0; t < 4; ++t) {
                const int col = nBase + t * 16 + ln;
                const int h = col >> 6, dk = col & 63;
                v8u16 pk;
#pragma unroll
                for (int r = 0; r < 8; ++r) pk[r] = f2bf(acc[mi][t][r]);
                *(v8u16*)(Out + ((size_t)(bb * H_ + h) * DK_ + dk) * S_ +
                          sTile + halfSel * 8) = pk;
            }
        } else {
            // Q/K with RoPE: partner element lives in lane^1 (adjacent d)
#pragma unroll
            for (int t = 0; t < 4; ++t) {
                const int col = nBase + t * 16 + ln;
                const int h = col >> 6, dk = col & 63;
                const int pairI = dk >> 1;
                const float invf = __expf(-0.2878231366f * (float)pairI); // 1e4^(-i/32)
#pragma unroll
                for (int r = 0; r < 8; ++r) {
                    float v  = acc[mi][t][r];
                    float pv = __shfl_xor(v, 1, 32);
                    const int s = sTile + r + halfSel * 8;
                    const float pos = (float)tp[s];
                    float ang = pos * invf, sn, cs;
                    __sincosf(ang, &sn, &cs);
                    float ov = (dk & 1) ? fmaf(cs, v,  sn * pv)
                                        : fmaf(cs, v, -sn * pv);
                    Out[((size_t)(bb * H_ + h) * S_ + s) * DK_ + dk] = f2bf(ov);
                }
            }
        }
    }
}

// ---------------------------------------------------------------------------
// K2: flash attention. One wave = one (b,h, 16-row q-tile). KV steps of 64.
//   scores = Q K^T (2 WMMA per 16x16 tile), online softmax in registers,
//   P converted C-layout -> A-layout through a private 2KB LDS slab,
//   O += P V using transposed V (contiguous B-fragments).
// ---------------------------------------------------------------------------
__global__ void mhsa_attn(const unsigned short* __restrict__ Qr,  // [BH][S][DK]
                          const unsigned short* __restrict__ Kr,  // [BH][S][DK]
                          const unsigned short* __restrict__ Vt,  // [BH][DK][S]
                          unsigned short*       __restrict__ AO)  // [B][S][D]
{
    __shared__ unsigned short Pl[8][16 * 64] __attribute__((aligned(16)));

    const int lane    = threadIdx.x & 31;
    const int wid     = threadIdx.x >> 5;
    const int gw      = blockIdx.x * 8 + wid;       // 0..4095
    const int bh      = gw >> 7;                    // 0..31
    const int qt      = gw & 127;
    const int qBase   = qt * 16;
    const int ln      = lane & 15;
    const int halfSel = lane >> 4;
    const int koffA   = halfSel * 16;
    const int koffB   = halfSel * 16;

    const unsigned short* Qbh = Qr + (size_t)bh * S_ * DK_;
    const unsigned short* Kbh = Kr + (size_t)bh * S_ * DK_;
    const unsigned short* Vbh = Vt + (size_t)bh * DK_ * S_;
    unsigned short* Pw = Pl[wid];

    // Q fragments (d = 0..31, 32..63)
    v16bf aQ[2];
    {
        const unsigned short* qrow = Qbh + (size_t)(qBase + ln) * DK_;
#pragma unroll
        for (int c = 0; c < 2; ++c) {
            v8bf lo = *(const v8bf*)(qrow + c * 32 + koffA);
            v8bf hi = *(const v8bf*)(qrow + c * 32 + koffA + 8);
            aQ[c] = concat8(lo, hi);
        }
    }

    v8f o[4] = {};
    float m_run[8], l_run[8];
#pragma unroll
    for (int r = 0; r < 8; ++r) { m_run[r] = -3.0e38f; l_run[r] = 0.0f; }

    const int numKv = (qBase + 16 + 63) >> 6;       // last tile end <= S

    for (int kt = 0; kt < numKv; ++kt) {
        const int kvBase = kt << 6;
        if (kvBase + 64 < S_)
            __builtin_prefetch(Kbh + (size_t)(kvBase + 64 + ln) * DK_, 0, 0);

        // ---- scores: 4 x (16x16) tiles over 64 kv columns ----
        v8f sc[4];
#pragma unroll
        for (int nt = 0; nt < 4; ++nt) {
            const unsigned short* krow = Kbh + (size_t)(kvBase + nt * 16 + ln) * DK_;
            v16bf b0 = *(const v16bf*)(krow + koffB);
            v16bf b1 = *(const v16bf*)(krow + 32 + koffB);
            v8f s = {};
            s = wmma_bf16(aQ[0], b0, s);
            s = wmma_bf16(aQ[1], b1, s);
            sc[nt] = s;
        }

        // ---- scale (1/sqrt(64)) + causal mask ----
#pragma unroll
        for (int nt = 0; nt < 4; ++nt) {
            const int kcol = kvBase + nt * 16 + ln;
#pragma unroll
            for (int r = 0; r < 8; ++r) {
                const int qrow_ = qBase + r + halfSel * 8;
                float v = sc[nt][r] * 0.125f;
                sc[nt][r] = (kcol <= qrow_) ? v : -1.0e30f;
            }
        }

        // ---- online softmax (per-row, 16-lane shuffle reductions) ----
        float corr[8];
#pragma unroll
        for (int r = 0; r < 8; ++r) {
            float m = sc[0][r];
#pragma unroll
            for (int t = 1; t < 4; ++t) m = fmaxf(m, sc[t][r]);
            m = rmax16(m);
            const float mn = fmaxf(m_run[r], m);
            const float cr = __expf(m_run[r] - mn);
            float rs = 0.0f;
#pragma unroll
            for (int t = 0; t < 4; ++t) {
                float p = __expf(sc[t][r] - mn);
                sc[t][r] = p;
                rs += p;
            }
            rs = rsum16(rs);
            l_run[r] = l_run[r] * cr + rs;
            m_run[r] = mn;
            corr[r]  = cr;
        }

        // rescale O accumulators
#pragma unroll
        for (int dt = 0; dt < 4; ++dt)
#pragma unroll
            for (int r = 0; r < 8; ++r) o[dt][r] *= corr[r];

        // ---- P: C-layout -> A-layout via private LDS slab (bf16) ----
#pragma unroll
        for (int nt = 0; nt < 4; ++nt)
#pragma unroll
            for (int r = 0; r < 8; ++r)
                Pw[(r + halfSel * 8) * 64 + nt * 16 + ln] = f2bf(sc[nt][r]);
        __builtin_amdgcn_wave_barrier();   // keep DS store->load order

        // ---- O += P x V ----
#pragma unroll
        for (int c = 0; c < 2; ++c) {
            const unsigned short* prow = Pw + ln * 64 + c * 32;
            v8bf lo = *(const v8bf*)(prow + koffA);
            v8bf hi = *(const v8bf*)(prow + koffA + 8);
            v16bf aP = concat8(lo, hi);
#pragma unroll
            for (int dt = 0; dt < 4; ++dt) {
                const unsigned short* vrow =
                    Vbh + (size_t)(dt * 16 + ln) * S_ + kvBase + c * 32 + koffB;
                v16bf bV = *(const v16bf*)vrow;
                o[dt] = wmma_bf16(aP, bV, o[dt]);
            }
        }
    }

    // ---- epilogue: AO[b][s][h*64 + d] (bf16) ----
    const int b = bh >> 4, h = bh & 15;
#pragma unroll
    for (int dt = 0; dt < 4; ++dt) {
#pragma unroll
        for (int r = 0; r < 8; ++r) {
            const int s = qBase + r + halfSel * 8;
            float ov = o[dt][r] / l_run[r];
            AO[((size_t)(b * S_ + s)) * D_ + h * DK_ + dt * 16 + ln] = f2bf(ov);
        }
    }
}

// ---------------------------------------------------------------------------
// K3: output projection  out[row,o] = sum_d AO[row,d] * Wo[o,d]  -> fp32
// Wave tile: 32 rows x 64 cols.
// ---------------------------------------------------------------------------
__global__ void mhsa_oproj(const unsigned short* __restrict__ AOb, // [4096][1024]
                           const unsigned short* __restrict__ Wb,  // [1024][1024]
                           float*                __restrict__ out) {
    const int lane    = threadIdx.x & 31;
    const int wid     = threadIdx.x >> 5;
    const int gw      = blockIdx.x * 8 + wid;       // 0..2047
    const int mBase   = (gw >> 4) * 32;
    const int nBase   = (gw & 15) * 64;
    const int ln      = lane & 15;
    const int halfSel = lane >> 4;
    const int koffA   = halfSel * 16;
    const int koffB   = halfSel * 16;

    v8f acc[2][4] = {};
    const unsigned short* xrow0 = AOb + (size_t)(mBase + ln) * D_;
    const unsigned short* xrow1 = AOb + (size_t)(mBase + 16 + ln) * D_;

    for (int k = 0; k < D_; k += 32) {
        if (k + 64 < D_) {
            __builtin_prefetch(xrow0 + k + 64, 0, 0);
            __builtin_prefetch(xrow1 + k + 64, 0, 0);
        }
        v16bf a0 = concat8(*(const v8bf*)(xrow0 + k + koffA),
                           *(const v8bf*)(xrow0 + k + koffA + 8));
        v16bf a1 = concat8(*(const v8bf*)(xrow1 + k + koffA),
                           *(const v8bf*)(xrow1 + k + koffA + 8));
#pragma unroll
        for (int t = 0; t < 4; ++t) {
            const unsigned short* wrow = Wb + (size_t)(nBase + t * 16 + ln) * D_;
            v16bf b = *(const v16bf*)(wrow + k + koffB);
            acc[0][t] = wmma_bf16(a0, b, acc[0][t]);
            acc[1][t] = wmma_bf16(a1, b, acc[1][t]);
        }
    }

#pragma unroll
    for (int mi = 0; mi < 2; ++mi) {
#pragma unroll
        for (int t = 0; t < 4; ++t) {
            const int col = nBase + t * 16 + ln;
#pragma unroll
            for (int r = 0; r < 8; ++r) {
                const int row = mBase + mi * 16 + r + halfSel * 8;
                out[(size_t)row * D_ + col] = acc[mi][t][r];
            }
        }
    }
}

// ---------------------------------------------------------------------------
// Host-side launcher
// ---------------------------------------------------------------------------
extern "C" void kernel_launch(void* const* d_in, const int* in_sizes, int n_in,
                              void* d_out, int out_size, void* d_ws, size_t ws_size,
                              hipStream_t stream) {
    (void)in_sizes; (void)n_in; (void)out_size; (void)ws_size;

    const float* x  = (const float*)d_in[0];
    const int*   tp = (const int*)d_in[1];
    const float* Wq = (const float*)d_in[2];
    const float* Wk = (const float*)d_in[3];
    const float* Wv = (const float*)d_in[4];
    const float* Wo = (const float*)d_in[5];
    float* out = (float*)d_out;

    char* ws = (char*)d_ws;
    const size_t MB = 1024 * 1024;
    unsigned short* xb = (unsigned short*)(ws + 0);        // 8 MB: x bf16 [4096][1024]
    unsigned short* wb = (unsigned short*)(ws + 8  * MB);  // 2 MB: current weight bf16
    unsigned short* Qr = (unsigned short*)(ws + 10 * MB);  // 8 MB: roped Q [BH][S][DK]
    unsigned short* Kr = (unsigned short*)(ws + 18 * MB);  // 8 MB: roped K [BH][S][DK]
    unsigned short* Vt = (unsigned short*)(ws + 26 * MB);  // 8 MB: V^T    [BH][DK][S]
    unsigned short* AO = xb;                               // reuse: attn out [B][S][D]

    const int nX = MROWS * D_;   // 4M
    const int nW = D_ * D_;      // 1M

    dim3 blk(256);
    mhsa_cvt_bf16<<<dim3(2048), blk, 0, stream>>>(x, xb, nX);

    // GEMM grid: 2048 wave-tiles (32x64) / 8 waves per block = 256 blocks
    dim3 ggrid(256);

    mhsa_cvt_bf16<<<dim3(512), blk, 0, stream>>>(Wq, wb, nW);
    mhsa_proj<<<ggrid, blk, 0, stream>>>(xb, wb, tp, Qr, /*mode=*/0);

    mhsa_cvt_bf16<<<dim3(512), blk, 0, stream>>>(Wk, wb, nW);
    mhsa_proj<<<ggrid, blk, 0, stream>>>(xb, wb, tp, Kr, /*mode=*/1);

    mhsa_cvt_bf16<<<dim3(512), blk, 0, stream>>>(Wv, wb, nW);
    mhsa_proj<<<ggrid, blk, 0, stream>>>(xb, wb, tp, Vt, /*mode=*/2);

    // attention: 32 (b,h) * 128 q-tiles = 4096 waves -> 512 blocks
    mhsa_attn<<<dim3(512), blk, 0, stream>>>(Qr, Kr, Vt, AO);

    mhsa_cvt_bf16<<<dim3(512), blk, 0, stream>>>(Wo, wb, nW);
    mhsa_oproj<<<ggrid, blk, 0, stream>>>(AO, wb, out);
}
